// BiLSTM_GAT_FC_4655744549120
// MI455X (gfx1250) — compile-verified, matrix operand order checked by reading, not run on hardware
//
#include <hip/hip_runtime.h>

// ---------------------------------------------------------------------------
// BiLSTM (WMMA-fused, async global->LDS weight staging) -> 2x GAT (WMMA
// projection + atomic segment softmax) -> FC (WMMA). gfx1250 / wave32.
// Matrix math on v_wmma_f32_16x16x32_f16; weight staging on
// global_load_async_to_lds_b128 (ASYNCcnt); x-row prefetch via
// global_prefetch_b8.
// ---------------------------------------------------------------------------

typedef _Float16 v8h  __attribute__((ext_vector_type(8)));
typedef _Float16 v16h __attribute__((ext_vector_type(16)));
typedef float    v8f  __attribute__((ext_vector_type(8)));

#define T_STEPS 20
#define FEAT    128
#define HID     128
#define LROW    136   // padded row stride in halfs (272B = 17*16B, LDS conflict-free)

// A/B fragment builder from f32 memory (rows contiguous in K):
// halves 0-7 <- p[0..7], halves 8-15 <- p[16..23]; caller passes p = row + kc*32 + hi8*8
__device__ __forceinline__ v16h frag_from_f32(const float* __restrict__ p) {
  v16h r;
#pragma unroll
  for (int i = 0; i < 8; ++i) { r[i] = (_Float16)p[i]; r[i + 8] = (_Float16)p[i + 16]; }
  return r;
}

// Same fragment pattern from f16 memory (LDS or global), two 16B vector loads.
__device__ __forceinline__ v16h frag_from_f16(const _Float16* p) {
  v8h lo = *(const v8h*)(p);
  v8h hi = *(const v8h*)(p + 16);
  v16h r;
#pragma unroll
  for (int i = 0; i < 8; ++i) { r[i] = lo[i]; r[i + 8] = hi[i]; }
  return r;
}

__device__ __forceinline__ v8f wmma16(v16h a, v16h b, v8f c) {
  return __builtin_amdgcn_wmma_f32_16x16x32_f16(false, a, false, b, (short)0, c,
                                                false, false);
}

// Async 16B copy global -> LDS, tracked by ASYNCcnt (no VGPR data round-trip).
// lds_off is the wave-relative LDS byte offset (low 32 bits of a flat shared
// pointer); gaddr is a 64-bit global address (VGPR pair).
__device__ __forceinline__ void async_copy_b128(unsigned lds_off, const void* gptr) {
  asm volatile("global_load_async_to_lds_b128 %0, %1, off"
               :: "v"(lds_off), "v"((unsigned long long)(size_t)gptr)
               : "memory");
}
__device__ __forceinline__ void wait_async_zero() {
  asm volatile("s_wait_asynccnt 0" ::: "memory");
}

// ---------------------------------------------------------------------------
// One-shot pack: f32 [512][128] weight -> padded f16 [512][LROW] image in ws,
// so the LSTM kernel can async-DMA it straight into LDS.
// ---------------------------------------------------------------------------
__global__ void pack_f16_pad_kernel(const float* __restrict__ src,
                                    _Float16* __restrict__ dst)
{
  int i = blockIdx.x * blockDim.x + threadIdx.x;
  if (i < 4 * HID * HID) {
    int r = i >> 7, c = i & (HID - 1);
    dst[r * LROW + c] = (_Float16)src[i];
  }
}

// ---------------------------------------------------------------------------
// Fused LSTM over one direction. Block = 16 rows x 8 waves (256 thr).
// Wave w owns gate cols [16w,16w+16) of each gate => h/c cols [16w,16w+16).
// Wih async-staged into LDS (f16, padded); Whh fragments resident in VGPRs.
// ---------------------------------------------------------------------------
__global__ __launch_bounds__(256)
void lstm_kernel(const float* __restrict__ x, const int* __restrict__ lengths,
                 const _Float16* __restrict__ wih_pad,  // [512][LROW] f16
                 const float* __restrict__ Whh,
                 const float* __restrict__ bias, _Float16* __restrict__ hcat,
                 int reverse)
{
  __shared__ __attribute__((aligned(16))) _Float16 wih_s[4 * HID * LROW]; // 139264 B
  __shared__ __attribute__((aligned(16))) _Float16 h_s[16 * LROW];        //   4352 B

  const int tid    = threadIdx.x;
  const int lane   = tid & 31;
  const int w      = tid >> 5;
  const int mbase  = blockIdx.x * 16;
  const int m      = lane & 15;
  const int hi8    = lane >> 4;     // 0: K {0..7,16..23}, 1: K {8..15,24..31}
  const int rowoff = hi8 * 8;       // C-matrix row offset for this half-wave
  const int k8     = hi8 * 8;

  // Async-stage the padded f16 Wih image into LDS (8704 x 16B chunks).
  {
    const int CHUNKS = (4 * HID * LROW * 2) / 16;
    const unsigned lds_base = (unsigned)(size_t)(&wih_s[0]);
    const char* gsrc = (const char*)wih_pad;
    for (int i = tid; i < CHUNKS; i += 256)
      async_copy_b128(lds_base + (unsigned)(i * 16), gsrc + (size_t)i * 16);
  }
  // Zero the h tile while the DMA is in flight.
  for (int i = tid; i < 16 * LROW; i += 256) h_s[i] = (_Float16)0.f;

  // Whh fragments resident in VGPRs: 4 gates x 4 k-chunks (overlaps with DMA).
  v16h whhf[4][4];
#pragma unroll
  for (int gI = 0; gI < 4; ++gI) {
#pragma unroll
    for (int kc = 0; kc < 4; ++kc) {
      const float* wp = Whh + (size_t)(gI * HID + w * 16 + m) * HID + kc * 32 + k8;
      whhf[gI][kc] = frag_from_f32(wp);
    }
  }

  const int myrow = mbase + m;          // row this lane supplies for A fragments
  const int mylen = lengths[myrow];
  int lenr[8];
#pragma unroll
  for (int r = 0; r < 8; ++r) lenr[r] = lengths[mbase + rowoff + r];
  float bgate[4];
#pragma unroll
  for (int gI = 0; gI < 4; ++gI) bgate[gI] = bias[gI * HID + w * 16 + m];

  float cst[8], hst[8];
#pragma unroll
  for (int r = 0; r < 8; ++r) { cst[r] = 0.f; hst[r] = 0.f; }

  wait_async_zero();   // Wih resident in LDS
  __syncthreads();

  for (int t = 0; t < T_STEPS; ++t) {
    // per-row time index (backward dir: clip(len-1-t, 0))
    int tix = reverse ? ((mylen - 1 - t) > 0 ? (mylen - 1 - t) : 0) : t;
    const float* xr = x + ((size_t)myrow * T_STEPS + tix) * FEAT;

    // Prefetch next step's x row (reversed-time pattern defeats HW stride pf).
    if (t + 1 < T_STEPS) {
      int tnx = reverse ? ((mylen - 2 - t) > 0 ? (mylen - 2 - t) : 0) : t + 1;
      __builtin_prefetch(x + ((size_t)myrow * T_STEPS + tnx) * FEAT, 0, 3);
    }

    v16h ax[4];
#pragma unroll
    for (int kc = 0; kc < 4; ++kc) ax[kc] = frag_from_f32(xr + kc * 32 + k8);

    v8f acc[4];
#pragma unroll
    for (int gI = 0; gI < 4; ++gI) {
      v8f a = {};
      const int wr = gI * HID + w * 16 + m;   // weight row == B column for this lane
#pragma unroll
      for (int kc = 0; kc < 4; ++kc) {
        const int k0 = kc * 32 + k8;
        v16h ah = frag_from_f16(&h_s[m * LROW + k0]);
        a = wmma16(ah, whhf[gI][kc], a);
        v16h bi = frag_from_f16(&wih_s[wr * LROW + k0]);
        a = wmma16(ax[kc], bi, a);
      }
      acc[gI] = a;
    }

    __syncthreads();   // everyone done reading h_s for this step
#pragma unroll
    for (int r = 0; r < 8; ++r) {
      float ig = 1.f / (1.f + __expf(-(acc[0][r] + bgate[0])));
      float fg = 1.f / (1.f + __expf(-(acc[1][r] + bgate[1])));
      float gg = tanhf(acc[2][r] + bgate[2]);
      float og = 1.f / (1.f + __expf(-(acc[3][r] + bgate[3])));
      float cn = fg * cst[r] + ig * gg;
      float hn = og * tanhf(cn);
      bool  mk = t < lenr[r];
      cst[r] = mk ? cn : cst[r];
      hst[r] = mk ? hn : hst[r];
      h_s[(rowoff + r) * LROW + w * 16 + m] = (_Float16)hst[r];
    }
    __syncthreads();   // new h visible before next step's reads
  }

  // last-valid hidden -> hcat [N][256], cols [dir*128 .. )
#pragma unroll
  for (int r = 0; r < 8; ++r) {
    hcat[(size_t)(mbase + rowoff + r) * (2 * HID) + reverse * HID + w * 16 + m] =
        (_Float16)hst[r];
  }
}

// ---------------------------------------------------------------------------
// Generic WMMA GEMM: C[M][NC] = A[M][K](f16) * B[NC][K](f32)^T (+bias).
// One wave per 16x16 output tile; NC-bounds guarded (used for FC with NC=10).
// ---------------------------------------------------------------------------
__global__ __launch_bounds__(32)
void gemm_wmma_kernel(const _Float16* __restrict__ A, const float* __restrict__ B,
                      const float* __restrict__ bias, float* __restrict__ C,
                      int M, int K, int NC)
{
  const int lane  = threadIdx.x;
  const int m     = lane & 15;
  const int hi8   = lane >> 4;
  const int k8    = hi8 * 8;
  const int mbase = blockIdx.x * 16;
  const int nbase = blockIdx.y * 16;
  const int ncol  = nbase + m;
  const bool bvalid = ncol < NC;

  const _Float16* ap = A + (size_t)(mbase + m) * K;
  const float*    bp = B + (size_t)(bvalid ? ncol : 0) * K;

  v8f acc = {};
  const int nk = K >> 5;
  for (int kc = 0; kc < nk; ++kc) {
    const int k0 = kc * 32 + k8;
    v16h af = frag_from_f16(ap + k0);
    v16h bf = frag_from_f32(bp + k0);
    if (!bvalid) {
#pragma unroll
      for (int i = 0; i < 16; ++i) bf[i] = (_Float16)0.f;
    }
    acc = wmma16(af, bf, acc);
  }

  const int rowoff = hi8 * 8;
  if (bvalid) {
    float bv = bias ? bias[ncol] : 0.f;
#pragma unroll
    for (int r = 0; r < 8; ++r)
      C[(size_t)(mbase + rowoff + r) * NC + ncol] = acc[r] + bv;
  }
}

// ---------------------------------------------------------------------------
// Attention logits: al_s/al_d [n, heads] = sum_g xh[n,h,g] * a_src/dst[h,g].
// One wave per node; GH == 128.
// ---------------------------------------------------------------------------
__global__ __launch_bounds__(256)
void attn_logits_kernel(const float* __restrict__ xh, const float* __restrict__ a_src,
                        const float* __restrict__ a_dst, float* __restrict__ al_s,
                        float* __restrict__ al_d, int n_nodes, int heads)
{
  const int lane = threadIdx.x & 31;
  const int node = (int)(((size_t)blockIdx.x * blockDim.x + threadIdx.x) >> 5);
  if (node >= n_nodes) return;
  const float* xr = xh + (size_t)node * heads * 128;
  for (int h = 0; h < heads; ++h) {
    float ps = 0.f, pd = 0.f;
#pragma unroll
    for (int j = 0; j < 4; ++j) {
      int c = h * 128 + lane + 32 * j;
      float v = xr[c];
      ps += v * a_src[c];
      pd += v * a_dst[c];
    }
#pragma unroll
    for (int off = 16; off > 0; off >>= 1) {
      ps += __shfl_xor(ps, off, 32);
      pd += __shfl_xor(pd, off, 32);
    }
    if (lane == 0) {
      al_s[(size_t)node * heads + h] = ps;
      al_d[(size_t)node * heads + h] = pd;
    }
  }
}

__global__ void zero_kernel(float* __restrict__ p, size_t nelem) {
  size_t stride = (size_t)gridDim.x * blockDim.x;
  for (size_t i = (size_t)blockIdx.x * blockDim.x + threadIdx.x; i < nelem; i += stride)
    p[i] = 0.f;
}

// Order-preserving uint encoding of float for atomic max.
__device__ __forceinline__ unsigned ordenc(float f) {
  unsigned u = __float_as_uint(f);
  return (u & 0x80000000u) ? ~u : (u | 0x80000000u);
}
__device__ __forceinline__ float orddec(unsigned e) {
  unsigned u = (e & 0x80000000u) ? (e ^ 0x80000000u) : ~e;
  return __uint_as_float(u);
}

__device__ __forceinline__ void edge_sd(const int* __restrict__ ei, size_t e, int E_,
                                        int& s, int& d) {
  if (e < (size_t)E_) { s = ei[e]; d = ei[(size_t)E_ + e]; }
  else                { s = d = (int)(e - (size_t)E_); }   // self-loop
}

__global__ void edge_max_kernel(const int* __restrict__ ei, const float* __restrict__ al_s,
                                const float* __restrict__ al_d, unsigned* __restrict__ emax,
                                int E_, int n_nodes, int heads)
{
  size_t total  = (size_t)(E_ + n_nodes) * heads;
  size_t stride = (size_t)gridDim.x * blockDim.x;
  for (size_t i = (size_t)blockIdx.x * blockDim.x + threadIdx.x; i < total; i += stride) {
    int h = (int)(i % heads);
    size_t e = i / heads;
    int s, d; edge_sd(ei, e, E_, s, d);
    float v = al_s[(size_t)s * heads + h] + al_d[(size_t)d * heads + h];
    v = v > 0.f ? v : 0.2f * v;   // leaky_relu(0.2)
    atomicMax(&emax[(size_t)d * heads + h], ordenc(v));
  }
}

__global__ void edge_exp_kernel(const int* __restrict__ ei, const float* __restrict__ al_s,
                                const float* __restrict__ al_d, const unsigned* __restrict__ emax,
                                float* __restrict__ ee, float* __restrict__ denom,
                                int E_, int n_nodes, int heads)
{
  size_t total  = (size_t)(E_ + n_nodes) * heads;
  size_t stride = (size_t)gridDim.x * blockDim.x;
  for (size_t i = (size_t)blockIdx.x * blockDim.x + threadIdx.x; i < total; i += stride) {
    int h = (int)(i % heads);
    size_t e = i / heads;
    int s, d; edge_sd(ei, e, E_, s, d);
    float v = al_s[(size_t)s * heads + h] + al_d[(size_t)d * heads + h];
    v = v > 0.f ? v : 0.2f * v;
    float ex = __expf(v - orddec(emax[(size_t)d * heads + h]));
    ee[i] = ex;
    atomicAdd(&denom[(size_t)d * heads + h], ex);
  }
}

// One wave per (edge, head): 128-dim alpha-weighted scatter-add.
__global__ __launch_bounds__(256)
void edge_agg_kernel(const int* __restrict__ ei, const float* __restrict__ xh,
                     const float* __restrict__ ee, const float* __restrict__ denom,
                     float* __restrict__ agg, int E_, int n_nodes, int heads)
{
  const int lane = threadIdx.x & 31;
  size_t wid   = ((size_t)blockIdx.x * blockDim.x + threadIdx.x) >> 5;
  size_t total = (size_t)(E_ + n_nodes) * heads;
  if (wid >= total) return;
  int h = (int)(wid % heads);
  size_t e = wid / heads;
  int s, d; edge_sd(ei, e, E_, s, d);
  float alpha = ee[wid] / (denom[(size_t)d * heads + h] + 1e-16f);
  const float* sf = xh  + ((size_t)s * heads + h) * 128;
  float*       df = agg + ((size_t)d * heads + h) * 128;
#pragma unroll
  for (int j = 0; j < 4; ++j) {
    int c = lane + 32 * j;
    atomicAdd(&df[c], sf[c] * alpha);
  }
}

__global__ void relu_bias_f16_kernel(const float* __restrict__ acc, const float* __restrict__ bias,
                                     _Float16* __restrict__ out, size_t nelem, int nc)
{
  size_t stride = (size_t)gridDim.x * blockDim.x;
  for (size_t i = (size_t)blockIdx.x * blockDim.x + threadIdx.x; i < nelem; i += stride) {
    float v = acc[i] + bias[i % nc];
    out[i] = (_Float16)(v > 0.f ? v : 0.f);
  }
}

// ---------------------------------------------------------------------------
extern "C" void kernel_launch(void* const* d_in, const int* in_sizes, int n_in,
                              void* d_out, int out_size, void* d_ws, size_t ws_size,
                              hipStream_t stream)
{
  const float* x       = (const float*)d_in[0];
  const int*   lengths = (const int*)  d_in[1];
  const int*   ei      = (const int*)  d_in[2];   // [2][E] flat: src then dst
  const float* Wih_f   = (const float*)d_in[3];
  const float* Whh_f   = (const float*)d_in[4];
  const float* b_f     = (const float*)d_in[5];
  const float* Wih_b   = (const float*)d_in[6];
  const float* Whh_b   = (const float*)d_in[7];
  const float* b_b     = (const float*)d_in[8];
  const float* W1      = (const float*)d_in[9];
  const float* a1_src  = (const float*)d_in[10];
  const float* a1_dst  = (const float*)d_in[11];
  const float* bias1   = (const float*)d_in[12];
  const float* W2      = (const float*)d_in[13];
  const float* a2_src  = (const float*)d_in[14];
  const float* a2_dst  = (const float*)d_in[15];
  const float* bias2   = (const float*)d_in[16];
  const float* Wfc     = (const float*)d_in[17];
  const float* bfc     = (const float*)d_in[18];

  const int n  = in_sizes[1];          // 20000 (multiple of 16)
  const int E_ = in_sizes[2] / 2;      // 640000
  const int EE = E_ + n;

  // ---- workspace layout (bump allocator, 256B aligned) ----
  char* wp = (char*)d_ws;
  auto alloc = [&](size_t bytes) -> char* {
    char* r = wp; wp += (bytes + 255) & ~(size_t)255; return r;
  };
  _Float16* wihp_f = (_Float16*)alloc((size_t)4 * HID * LROW * 2);
  _Float16* wihp_b = (_Float16*)alloc((size_t)4 * HID * LROW * 2);
  _Float16* hcat  = (_Float16*)alloc((size_t)n * 256 * 2);
  float*    xh1   = (float*)   alloc((size_t)n * 512 * 4);
  float*    al1s  = (float*)   alloc((size_t)n * 4 * 4);
  float*    al1d  = (float*)   alloc((size_t)n * 4 * 4);
  unsigned* emax1 = (unsigned*)alloc((size_t)n * 4 * 4);
  float*    den1  = (float*)   alloc((size_t)n * 4 * 4);
  float*    agg1  = (float*)   alloc((size_t)n * 512 * 4);
  float*    ee1   = (float*)   alloc((size_t)EE * 4 * 4);
  _Float16* h1    = (_Float16*)alloc((size_t)n * 512 * 2);
  float*    xh2   = (float*)   alloc((size_t)n * 128 * 4);
  float*    al2s  = (float*)   alloc((size_t)n * 4);
  float*    al2d  = (float*)   alloc((size_t)n * 4);
  unsigned* emax2 = (unsigned*)alloc((size_t)n * 4);
  float*    den2  = (float*)   alloc((size_t)n * 4);
  float*    agg2  = (float*)   alloc((size_t)n * 128 * 4);
  float*    ee2   = (float*)   alloc((size_t)EE * 4);
  _Float16* h2    = (_Float16*)alloc((size_t)n * 128 * 2);

  const int mt = n / 16;   // 1250 row tiles

  // ---- pack Wih into padded f16 images (async-DMA source) ----
  pack_f16_pad_kernel<<<(4 * HID * HID + 255) / 256, 256, 0, stream>>>(Wih_f, wihp_f);
  pack_f16_pad_kernel<<<(4 * HID * HID + 255) / 256, 256, 0, stream>>>(Wih_b, wihp_b);

  // ---- BiLSTM (fused input + recurrent GEMMs, WMMA) ----
  lstm_kernel<<<mt, 256, 0, stream>>>(x, lengths, wihp_f, Whh_f, b_f, hcat, 0);
  lstm_kernel<<<mt, 256, 0, stream>>>(x, lengths, wihp_b, Whh_b, b_b, hcat, 1);

  // ---- GAT layer 1 ----
  gemm_wmma_kernel<<<dim3(mt, 32), 32, 0, stream>>>(hcat, W1, nullptr, xh1, n, 256, 512);
  attn_logits_kernel<<<(n + 7) / 8, 256, 0, stream>>>(xh1, a1_src, a1_dst, al1s, al1d, n, 4);
  zero_kernel<<<2048, 256, 0, stream>>>((float*)emax1, (size_t)n * 4);
  zero_kernel<<<2048, 256, 0, stream>>>(den1, (size_t)n * 4);
  zero_kernel<<<4096, 256, 0, stream>>>(agg1, (size_t)n * 512);
  {
    size_t tot = (size_t)EE * 4;
    int blk = (int)((tot + 255) / 256);
    edge_max_kernel<<<blk, 256, 0, stream>>>(ei, al1s, al1d, emax1, E_, n, 4);
    edge_exp_kernel<<<blk, 256, 0, stream>>>(ei, al1s, al1d, emax1, ee1, den1, E_, n, 4);
    int ablk = (int)((tot + 7) / 8);
    edge_agg_kernel<<<ablk, 256, 0, stream>>>(ei, xh1, ee1, den1, agg1, E_, n, 4);
  }
  relu_bias_f16_kernel<<<4096, 256, 0, stream>>>(agg1, bias1, h1, (size_t)n * 512, 512);

  // ---- GAT layer 2 (heads = 1) ----
  gemm_wmma_kernel<<<dim3(mt, 8), 32, 0, stream>>>(h1, W2, nullptr, xh2, n, 512, 128);
  attn_logits_kernel<<<(n + 7) / 8, 256, 0, stream>>>(xh2, a2_src, a2_dst, al2s, al2d, n, 1);
  zero_kernel<<<512, 256, 0, stream>>>((float*)emax2, (size_t)n);
  zero_kernel<<<512, 256, 0, stream>>>(den2, (size_t)n);
  zero_kernel<<<4096, 256, 0, stream>>>(agg2, (size_t)n * 128);
  {
    size_t tot = (size_t)EE;
    int blk = (int)((tot + 255) / 256);
    edge_max_kernel<<<blk, 256, 0, stream>>>(ei, al2s, al2d, emax2, E_, n, 1);
    edge_exp_kernel<<<blk, 256, 0, stream>>>(ei, al2s, al2d, emax2, ee2, den2, E_, n, 1);
    int ablk = (int)((tot + 7) / 8);
    edge_agg_kernel<<<ablk, 256, 0, stream>>>(ei, xh2, ee2, den2, agg2, E_, n, 1);
  }
  relu_bias_f16_kernel<<<4096, 256, 0, stream>>>(agg2, bias2, h2, (size_t)n * 128, 128);

  // ---- FC: out = h2 @ Wfc^T + bfc  (NC = 10 padded to one 16-wide tile) ----
  gemm_wmma_kernel<<<dim3(mt, 1), 32, 0, stream>>>(h2, Wfc, bfc, (float*)d_out, n, 128, 10);
}
